// add_transformer_48584670053216
// MI455X (gfx1250) — matrix operand order repaired and checked by reading, not compiled, not run on hardware
//
#include <hip/hip_runtime.h>

// ---------------- problem constants (from reference setup_inputs) ----------------
#define BB    4
#define SQ    2048
#define SKK   2560          // Sq + cache
#define DD    512
#define CC    512
#define TT    2044          // Sq - KSIZE + 1
#define KS    5
#define SKP   2592          // padded kT row stride (SKK + 32) for band-edge overreads
#define QPAD  16            // extra q rows so conv window reads stay in-bounds
#define HROWS (BB*TT)       // 8176 (exactly 511 row-tiles of 16)

typedef __bf16 bf16_t;
typedef bf16_t v16bf __attribute__((ext_vector_type(16)));
typedef float  v8f   __attribute__((ext_vector_type(8)));
typedef unsigned int u32x4 __attribute__((ext_vector_type(4)));

union Frag {
  v16bf bf;
  u32x4 q[2];
  unsigned short s[16];
};

__device__ __forceinline__ unsigned short f2bf(float x) {
  union { float f; unsigned int u; } v; v.f = x;
  unsigned int r = v.u + 0x7FFFu + ((v.u >> 16) & 1u);   // round-to-nearest-even
  return (unsigned short)(r >> 16);
}

__device__ __forceinline__ v8f wmma_bf16(v16bf a, v16bf b, v8f c) {
  // D = A(16x32 bf16) x B(32x16 bf16) + C(16x16 f32)
  return __builtin_amdgcn_wmma_f32_16x16x32_bf16(false, a, false, b, (short)0, c, false, false);
}

// A-fragment: p = row_base + m*lda + kbA (kbA = lane<16 ? 0 : 8), then + k0.
__device__ __forceinline__ v16bf frag_ld_a(const unsigned short* p) {
  Frag f;
  f.q[0] = *reinterpret_cast<const u32x4*>(p);        // K kb..kb+7
  f.q[1] = *reinterpret_cast<const u32x4*>(p + 16);   // K kb+16..kb+23
  return f.bf;
}
// B-fragment: p = W + (n0+n)*ldw + kbB (kbB = lane<16 ? 0 : 8), then + k0.
__device__ __forceinline__ v16bf frag_ld_b(const unsigned short* p) {
  Frag f;
  f.q[0] = *reinterpret_cast<const u32x4*>(p);
  f.q[1] = *reinterpret_cast<const u32x4*>(p + 16);
  return f.bf;
}

// ---------------- small utility kernels ----------------
__global__ void zero_kernel(u32x4* p, size_t n4) {
  size_t i = blockIdx.x * (size_t)blockDim.x + threadIdx.x;
  size_t st = gridDim.x * (size_t)blockDim.x;
  u32x4 z; z[0] = 0; z[1] = 0; z[2] = 0; z[3] = 0;
  for (; i < n4; i += st) p[i] = z;
}

__global__ void cvt_bf16_kernel(const float* __restrict__ in, unsigned short* __restrict__ out, int n) {
  int i = blockIdx.x * blockDim.x + threadIdx.x;
  if (i < n) out[i] = f2bf(in[i]);
}

// conv_w[C][D][5] f32 -> 5 contiguous bf16 matrices W_j[C][D]
__global__ void convw_split_kernel(const float* __restrict__ cw, unsigned short* __restrict__ out) {
  int i = blockIdx.x * blockDim.x + threadIdx.x;
  if (i >= CC * DD * KS) return;
  int j  = i % KS;
  int cd = i / KS;
  out[(size_t)j * (CC * DD) + cd] = f2bf(cw[i]);
}

// row-wise layernorm over D=512, output bf16
__global__ void ln_bf16_kernel(const float* __restrict__ x, const float* __restrict__ g,
                               const float* __restrict__ b, unsigned short* __restrict__ y) {
  int row = blockIdx.x;
  int tid = threadIdx.x;                     // 256 threads, 2 elems each
  const float* xr = x + (size_t)row * DD;
  float x0 = xr[tid], x1 = xr[tid + 256];
  __shared__ float s1[256], s2[256];
  s1[tid] = x0 + x1;
  s2[tid] = x0 * x0 + x1 * x1;
  __syncthreads();
  for (int st = 128; st > 0; st >>= 1) {
    if (tid < st) { s1[tid] += s1[tid + st]; s2[tid] += s2[tid + st]; }
    __syncthreads();
  }
  float mean = s1[0] * (1.0f / DD);
  float var  = s2[0] * (1.0f / DD) - mean * mean;
  float inv  = rsqrtf(var + 1e-5f);
  unsigned short* yr = y + (size_t)row * DD;
  yr[tid]       = f2bf((x0 - mean) * inv * g[tid]       + b[tid]);
  yr[tid + 256] = f2bf((x1 - mean) * inv * g[tid + 256] + b[tid + 256]);
}

// ---- stage 1: k = relu(lnk@Wk^T+bk) * (lnv@Wv^T+bv), stored transposed kT[b][d][s] ----
// 16x32 tile of BOTH sub-GEMMs per wave; copy-free ping-pong pipelining (2 k-steps/iter).
__global__ void kv_gemm_kernel(const unsigned short* __restrict__ lnk,
                               const unsigned short* __restrict__ lnv,
                               const unsigned short* __restrict__ wk,
                               const unsigned short* __restrict__ wv,
                               const float* __restrict__ bk, const float* __restrict__ bv,
                               unsigned short* __restrict__ kT) {
  int m0  = blockIdx.x * 16;                // global row over B*Sk (tiles never straddle batches)
  int nt0 = blockIdx.y * 32;
  int lane = threadIdx.x;
  int m = lane & 15, kbA = (lane < 16) ? 0 : 8;
  int n = lane & 15, kbB = (lane < 16) ? 0 : 8;
  const unsigned short* pAk = lnk + (size_t)(m0 + m) * DD + kbA;
  const unsigned short* pAv = lnv + (size_t)(m0 + m) * DD + kbA;
  const unsigned short* pBk[2]; const unsigned short* pBv[2];
#pragma unroll
  for (int u = 0; u < 2; ++u) {
    pBk[u] = wk + (size_t)(nt0 + u * 16 + n) * DD + kbB;
    pBv[u] = wv + (size_t)(nt0 + u * 16 + n) * DD + kbB;
  }
  v8f ca[2] = {}; v8f cb[2] = {};
  v16bf ak0, av0, ak1, av1, bk0[2], bv0[2], bk1[2], bv1[2];
  ak0 = frag_ld_a(pAk); av0 = frag_ld_a(pAv);
#pragma unroll
  for (int u = 0; u < 2; ++u) { bk0[u] = frag_ld_b(pBk[u]); bv0[u] = frag_ld_b(pBv[u]); }
  for (int p = 0; p < 8; ++p) {
    int k1 = (2 * p + 1) << 5;
    ak1 = frag_ld_a(pAk + k1); av1 = frag_ld_a(pAv + k1);
#pragma unroll
    for (int u = 0; u < 2; ++u) { bk1[u] = frag_ld_b(pBk[u] + k1); bv1[u] = frag_ld_b(pBv[u] + k1); }
#pragma unroll
    for (int u = 0; u < 2; ++u) {
      ca[u] = wmma_bf16(ak0, bk0[u], ca[u]);
      cb[u] = wmma_bf16(av0, bv0[u], cb[u]);
    }
    int k2 = ((2 * p + 2) & 15) << 5;       // wraps to 0 on the final (unused) prefetch
    ak0 = frag_ld_a(pAk + k2); av0 = frag_ld_a(pAv + k2);
#pragma unroll
    for (int u = 0; u < 2; ++u) { bk0[u] = frag_ld_b(pBk[u] + k2); bv0[u] = frag_ld_b(pBv[u] + k2); }
#pragma unroll
    for (int u = 0; u < 2; ++u) {
      ca[u] = wmma_bf16(ak1, bk1[u], ca[u]);
      cb[u] = wmma_bf16(av1, bv1[u], cb[u]);
    }
  }
  int b    = m0 / SKK;
  int srow = m0 % SKK;
  int soff = (lane < 16) ? 0 : 8;           // C-layout: lane<16 -> m=r, lane>=16 -> m=8+r
#pragma unroll
  for (int u = 0; u < 2; ++u) {
    int n0 = nt0 + u * 16;
    float bkv = bk[n0 + n], bvv = bv[n0 + n];
    unsigned short hv[8];
#pragma unroll
    for (int r = 0; r < 8; ++r) {
      float a = ca[u][r] + bkv; a = a > 0.f ? a : 0.f;
      hv[r] = f2bf(a * (cb[u][r] + bvv));
    }
    u32x4 pk;
#pragma unroll
    for (int i = 0; i < 4; ++i)
      pk[i] = (unsigned int)hv[2 * i] | ((unsigned int)hv[2 * i + 1] << 16);
    // transposed store: row d = n0+n of kT, 8 consecutive s values -> one 16B store
    unsigned short* dst = kT + ((size_t)b * DD + (n0 + n)) * SKP + srow + soff;
    *reinterpret_cast<u32x4*>(dst) = pk;
  }
}

// ---- stage 2: q = lnq@Wq^T + bq; 16x64 per wave, copy-free ping-pong ----
__global__ void qproj_kernel(const unsigned short* __restrict__ lnq,
                             const unsigned short* __restrict__ wq,
                             const float* __restrict__ bq,
                             unsigned short* __restrict__ q) {
  int m0  = blockIdx.x * 16;
  int nt0 = blockIdx.y * 64;
  int lane = threadIdx.x;
  int m = lane & 15, kbA = (lane < 16) ? 0 : 8;
  int n = lane & 15, kbB = (lane < 16) ? 0 : 8;
  const unsigned short* pA = lnq + (size_t)(m0 + m) * DD + kbA;
  const unsigned short* pB[4];
#pragma unroll
  for (int u = 0; u < 4; ++u) pB[u] = wq + (size_t)(nt0 + u * 16 + n) * DD + kbB;
  v8f acc[4] = {};
  v16bf a0, a1, b0[4], b1[4];
  a0 = frag_ld_a(pA);
#pragma unroll
  for (int u = 0; u < 4; ++u) b0[u] = frag_ld_b(pB[u]);
  for (int p = 0; p < 8; ++p) {
    int k1 = (2 * p + 1) << 5;
    a1 = frag_ld_a(pA + k1);
#pragma unroll
    for (int u = 0; u < 4; ++u) b1[u] = frag_ld_b(pB[u] + k1);
#pragma unroll
    for (int u = 0; u < 4; ++u) acc[u] = wmma_bf16(a0, b0[u], acc[u]);
    int k2 = ((2 * p + 2) & 15) << 5;
    a0 = frag_ld_a(pA + k2);
#pragma unroll
    for (int u = 0; u < 4; ++u) b0[u] = frag_ld_b(pB[u] + k2);
#pragma unroll
    for (int u = 0; u < 4; ++u) acc[u] = wmma_bf16(a1, b1[u], acc[u]);
  }
  int mb = (lane < 16) ? 0 : 8;
#pragma unroll
  for (int u = 0; u < 4; ++u) {
    int n0 = nt0 + u * 16;
    float bias = bq[n0 + n];
#pragma unroll
    for (int r = 0; r < 8; ++r)
      q[(size_t)(m0 + mb + r) * DD + n0 + n] = f2bf(acc[u][r] + bias);
  }
}

// ---- stage 3: conv(K=5) as one flattened 80-step WMMA chain; h = relu(.+cb); 16x64/wave ----
__global__ void conv_gemm_kernel(const unsigned short* __restrict__ q,
                                 const unsigned short* __restrict__ cw16,
                                 const float* __restrict__ cb,
                                 unsigned short* __restrict__ h) {
  int t0  = blockIdx.x * 16;
  int nt0 = blockIdx.y * 64;
  int b   = blockIdx.z;
  int lane = threadIdx.x;
  int m = lane & 15, kbA = (lane < 16) ? 0 : 8;
  int n = lane & 15, kbB = (lane < 16) ? 0 : 8;
  const unsigned short* pA = q + (size_t)(b * SQ + t0 + m) * DD + kbA;        // +j*DD per tap
  const unsigned short* pB[4];
#pragma unroll
  for (int u = 0; u < 4; ++u) pB[u] = cw16 + (size_t)(nt0 + u * 16 + n) * DD + kbB; // +j*CC*DD per tap
  v8f acc[4] = {};
  v16bf a0, a1, b0[4], b1[4];
  a0 = frag_ld_a(pA);
#pragma unroll
  for (int u = 0; u < 4; ++u) b0[u] = frag_ld_b(pB[u]);
  for (int p = 0; p < KS * 8; ++p) {        // 40 iters x 2 steps = 80 k-steps
    int i1 = 2 * p + 1;
    int oA1 = (i1 >> 4) * DD + ((i1 & 15) << 5);
    int oB1 = (i1 >> 4) * (CC * DD) + ((i1 & 15) << 5);
    a1 = frag_ld_a(pA + oA1);
#pragma unroll
    for (int u = 0; u < 4; ++u) b1[u] = frag_ld_b(pB[u] + oB1);
#pragma unroll
    for (int u = 0; u < 4; ++u) acc[u] = wmma_bf16(a0, b0[u], acc[u]);
    int i2 = 2 * p + 2; i2 = (i2 < KS * 16) ? i2 : 0;   // wrap final prefetch
    int oA2 = (i2 >> 4) * DD + ((i2 & 15) << 5);
    int oB2 = (i2 >> 4) * (CC * DD) + ((i2 & 15) << 5);
    a0 = frag_ld_a(pA + oA2);
#pragma unroll
    for (int u = 0; u < 4; ++u) b0[u] = frag_ld_b(pB[u] + oB2);
#pragma unroll
    for (int u = 0; u < 4; ++u) acc[u] = wmma_bf16(a1, b1[u], acc[u]);
  }
  int mb = (lane < 16) ? 0 : 8;
#pragma unroll
  for (int u = 0; u < 4; ++u) {
    int n0 = nt0 + u * 16;
    float bias = cb[n0 + n];
#pragma unroll
    for (int r = 0; r < 8; ++r) {
      int t = t0 + mb + r;
      if (t < TT) {
        float v = acc[u][r] + bias; v = v > 0.f ? v : 0.f;
        h[(size_t)(b * TT + t) * CC + n0 + n] = f2bf(v);
      }
    }
  }
}

// ---- stage 4: w = sigmoid(h@Wf^T + fb); 16x64 per wave, copy-free ping-pong ----
__global__ void fnn_gemm_kernel(const unsigned short* __restrict__ h,
                                const unsigned short* __restrict__ wf,
                                const float* __restrict__ fb,
                                unsigned short* __restrict__ wg) {
  int m0  = blockIdx.x * 16;                // rows over B*T (8176 = 511 exact tiles)
  int nt0 = blockIdx.y * 64;
  int lane = threadIdx.x;
  int m = lane & 15, kbA = (lane < 16) ? 0 : 8;
  int n = lane & 15, kbB = (lane < 16) ? 0 : 8;
  const unsigned short* pA = h + (size_t)(m0 + m) * CC + kbA;
  const unsigned short* pB[4];
#pragma unroll
  for (int u = 0; u < 4; ++u) pB[u] = wf + (size_t)(nt0 + u * 16 + n) * CC + kbB;
  v8f acc[4] = {};
  v16bf a0, a1, b0[4], b1[4];
  a0 = frag_ld_a(pA);
#pragma unroll
  for (int u = 0; u < 4; ++u) b0[u] = frag_ld_b(pB[u]);
  for (int p = 0; p < 8; ++p) {
    int k1 = (2 * p + 1) << 5;
    a1 = frag_ld_a(pA + k1);
#pragma unroll
    for (int u = 0; u < 4; ++u) b1[u] = frag_ld_b(pB[u] + k1);
#pragma unroll
    for (int u = 0; u < 4; ++u) acc[u] = wmma_bf16(a0, b0[u], acc[u]);
    int k2 = ((2 * p + 2) & 15) << 5;
    a0 = frag_ld_a(pA + k2);
#pragma unroll
    for (int u = 0; u < 4; ++u) b0[u] = frag_ld_b(pB[u] + k2);
#pragma unroll
    for (int u = 0; u < 4; ++u) acc[u] = wmma_bf16(a1, b1[u], acc[u]);
  }
  int mb = (lane < 16) ? 0 : 8;
#pragma unroll
  for (int u = 0; u < 4; ++u) {
    int n0 = nt0 + u * 16;
    float bias = fb[n0 + n];
#pragma unroll
    for (int r = 0; r < 8; ++r) {
      float x = acc[u][r] + bias;
      wg[(size_t)(m0 + mb + r) * CC + n0 + n] = f2bf(1.0f / (1.0f + __expf(-x)));
    }
  }
}

// ---- stage 5: banded GEMM out[t,d] = (1/C) sum_s w[t,s-t] * k[s,d] ----
// 128 threads = 4 waves share one t-tile's zero-padded w rows in LDS; 4 d-tiles/wave.
// Trip count is exactly 17 s-blocks for every tile: 8 ping-pong pairs + 1 tail.
__global__ void window_gemm_kernel(const unsigned short* __restrict__ wg,
                                   const unsigned short* __restrict__ kT,
                                   float* __restrict__ out) {
  __shared__ unsigned short lw[16][640];    // 64 zero-pad | 512 w row | 64 zero-pad
  int t0 = blockIdx.x * 16;
  int b  = blockIdx.z;
  int tid = threadIdx.x;
  for (int i = tid; i < 16 * 640; i += 128) {
    int mm = i / 640, c = i % 640;
    int t = t0 + mm, j = c - 64;
    unsigned short v = 0;
    if (j >= 0 && j < CC && t < TT)
      v = wg[(size_t)(b * TT + t) * CC + j];
    lw[mm][c] = v;
  }
  __syncthreads();

  int wv = tid >> 5, lane = tid & 31;
  int m   = lane & 15;
  int kbA = (lane < 16) ? 0 : 8;
  int n   = lane & 15;
  int kbB = (lane < 16) ? 0 : 8;
  const unsigned short* pB[4];
#pragma unroll
  for (int u = 0; u < 4; ++u) {
    int n0 = ((blockIdx.y * 4 + wv) * 4 + u) * 16;   // d-tile
    pB[u] = kT + (size_t)b * DD * SKP + (size_t)(n0 + n) * SKP + kbB;
  }
  v8f acc[4] = {};
  int sBeg = t0 & ~31;

  auto gatherA = [&](Frag& a, int s0) {
    int base = s0 + kbA - (t0 + m) + 64;    // LDS column (always in [0,640))
#pragma unroll
    for (int e = 0; e < 8; ++e) a.s[e]     = lw[m][base + e];
#pragma unroll
    for (int e = 0; e < 8; ++e) a.s[8 + e] = lw[m][base + 16 + e];
  };
  auto loadB = [&](Frag* bf, int s0) {
#pragma unroll
    for (int u = 0; u < 4; ++u) {
      const unsigned short* p = pB[u] + s0;
      bf[u].q[0] = *reinterpret_cast<const u32x4*>(p);
      bf[u].q[1] = *reinterpret_cast<const u32x4*>(p + 16);
    }
  };

  Frag a0, a1, b0[4], b1[4];
  gatherA(a0, sBeg);
  loadB(b0, sBeg);
  for (int p = 0; p < 8; ++p) {
    int s1 = sBeg + ((2 * p + 1) << 5);
    gatherA(a1, s1);
    loadB(b1, s1);
#pragma unroll
    for (int u = 0; u < 4; ++u) acc[u] = wmma_bf16(a0.bf, b0[u].bf, acc[u]);
    int s2 = sBeg + ((2 * p + 2) << 5);     // 2p+2 <= 16: always a valid block
    gatherA(a0, s2);
    loadB(b0, s2);
#pragma unroll
    for (int u = 0; u < 4; ++u) acc[u] = wmma_bf16(a1.bf, b1[u].bf, acc[u]);
  }
#pragma unroll
  for (int u = 0; u < 4; ++u) acc[u] = wmma_bf16(a0.bf, b0[u].bf, acc[u]);  // block 16 tail

  int mb = (lane < 16) ? 0 : 8;
  const float scale = 1.0f / (float)CC;
#pragma unroll
  for (int u = 0; u < 4; ++u) {
    int n0 = ((blockIdx.y * 4 + wv) * 4 + u) * 16;
#pragma unroll
    for (int r = 0; r < 8; ++r) {
      int t = t0 + mb + r;
      if (t < TT) out[(size_t)(b * TT + t) * DD + n0 + n] = acc[u][r] * scale;
    }
  }
}

// ---------------- host side ----------------
extern "C" void kernel_launch(void* const* d_in, const int* in_sizes, int n_in,
                              void* d_out, int out_size, void* d_ws, size_t ws_size,
                              hipStream_t stream) {
  (void)in_sizes; (void)n_in; (void)out_size; (void)ws_size;
  const float* query  = (const float*)d_in[0];
  const float* key    = (const float*)d_in[1];
  const float* value  = (const float*)d_in[2];
  const float* lnq_g  = (const float*)d_in[3];
  const float* lnq_b  = (const float*)d_in[4];
  const float* lnk_g  = (const float*)d_in[5];
  const float* lnk_b  = (const float*)d_in[6];
  const float* lnv_g  = (const float*)d_in[7];
  const float* lnv_b  = (const float*)d_in[8];
  const float* wq_w   = (const float*)d_in[9];
  const float* wq_b   = (const float*)d_in[10];
  const float* wk_w   = (const float*)d_in[11];
  const float* wk_b   = (const float*)d_in[12];
  const float* wv_w   = (const float*)d_in[13];
  const float* wv_b   = (const float*)d_in[14];
  const float* conv_w = (const float*)d_in[15];
  const float* conv_b = (const float*)d_in[16];
  const float* fnn_w  = (const float*)d_in[17];
  const float* fnn_b  = (const float*)d_in[18];
  float* out = (float*)d_out;

  // bump allocator over d_ws (~70 MB total, all bf16 scratch)
  char* ws = (char*)d_ws;
  size_t off = 0;
  auto alloc = [&](size_t bytes) -> void* {
    void* p = ws + off;
    off = (off + bytes + 255) & ~(size_t)255;
    return p;
  };
  unsigned short* lnq  = (unsigned short*)alloc(2ull * BB * SQ  * DD);
  unsigned short* lnk  = (unsigned short*)alloc(2ull * BB * SKK * DD);
  unsigned short* lnv  = (unsigned short*)alloc(2ull * BB * SKK * DD);
  unsigned short* wq16 = (unsigned short*)alloc(2ull * DD * DD);
  unsigned short* wk16 = (unsigned short*)alloc(2ull * DD * DD);
  unsigned short* wv16 = (unsigned short*)alloc(2ull * DD * DD);
  unsigned short* wf16 = (unsigned short*)alloc(2ull * CC * CC);
  unsigned short* cw16 = (unsigned short*)alloc(2ull * KS * CC * DD);
  unsigned short* qbuf = (unsigned short*)alloc(2ull * (BB * SQ + QPAD) * DD);
  unsigned short* kT   = (unsigned short*)alloc(2ull * BB * DD * SKP);
  unsigned short* hbuf = (unsigned short*)alloc(2ull * HROWS * CC);
  unsigned short* wg   = (unsigned short*)alloc(2ull * HROWS * CC);

  const int W2 = DD * DD;
  // weight conversions
  cvt_bf16_kernel<<<(W2 + 255) / 256, 256, 0, stream>>>(wq_w, wq16, W2);
  cvt_bf16_kernel<<<(W2 + 255) / 256, 256, 0, stream>>>(wk_w, wk16, W2);
  cvt_bf16_kernel<<<(W2 + 255) / 256, 256, 0, stream>>>(wv_w, wv16, W2);
  cvt_bf16_kernel<<<(W2 + 255) / 256, 256, 0, stream>>>(fnn_w, wf16, W2);
  convw_split_kernel<<<(CC * DD * KS + 255) / 256, 256, 0, stream>>>(conv_w, cw16);

  // zero padded scratch regions (q pad rows, kT band-edge pad) so overreads are finite zeros
  zero_kernel<<<2048, 256, 0, stream>>>((u32x4*)qbuf, (2ull * (BB * SQ + QPAD) * DD) / 16);
  zero_kernel<<<2048, 256, 0, stream>>>((u32x4*)kT,   (2ull * BB * DD * SKP) / 16);

  // layernorms
  ln_bf16_kernel<<<BB * SQ,  256, 0, stream>>>(query, lnq_g, lnq_b, lnq);
  ln_bf16_kernel<<<BB * SKK, 256, 0, stream>>>(key,   lnk_g, lnk_b, lnk);
  ln_bf16_kernel<<<BB * SKK, 256, 0, stream>>>(value, lnv_g, lnv_b, lnv);

  // stage 1: gated k/v combine -> kT (transposed, bf16)
  kv_gemm_kernel<<<dim3(BB * SKK / 16, DD / 32), 32, 0, stream>>>(lnk, lnv, wk16, wv16, wk_b, wv_b, kT);
  // stage 2: q projection
  qproj_kernel<<<dim3(BB * SQ / 16, DD / 64), 32, 0, stream>>>(lnq, wq16, wq_b, qbuf);
  // stage 3: conv as one flattened WMMA chain
  conv_gemm_kernel<<<dim3(SQ / 16, CC / 64, BB), 32, 0, stream>>>(qbuf, cw16, conv_b, hbuf);
  // stage 4: hypernet FNN + sigmoid
  fnn_gemm_kernel<<<dim3(HROWS / 16, CC / 64), 32, 0, stream>>>(hbuf, wf16, fnn_b, wg);
  // stage 5: banded window combine (WMMA over the 512-wide band only)
  window_gemm_kernel<<<dim3(SQ / 16, 2, BB), 128, 0, stream>>>(wg, kT, out);
}
